// NeRFNetwork_59528246723040
// MI455X (gfx1250) — matrix operand order repaired and checked by reading, not compiled, not run on hardware
//
#include <hip/hip_runtime.h>
#include <math.h>

#define NRAYS   4096
#define NSTEPS  64
#define NPTS    (NRAYS * NSTEPS)
#define TABLE_SIZE (1u << 19)
#define BOUND   1.0f
#define NFRAGS  36
#define WLDS_HALFS (NFRAGS * 512)   // 18432 halfs = 36864 B

typedef __attribute__((ext_vector_type(16))) _Float16 v16h;
typedef __attribute__((ext_vector_type(8)))  _Float16 h8;
typedef __attribute__((ext_vector_type(8)))  float    v8f;

// ---------------------------------------------------------------------------
// Build an A fragment (16x32 f16, one K-tile) from a row-major f16 row.
// Lane L holds row M=L%16, group g=L/16:
//   VGPR0-3  : K = g*8 + 0..7        (contiguous 16B)
//   VGPR4-7  : K = 16 + g*8 + 0..7   (contiguous 16B)
// `p` must point at &row[kt*32 + g*8].
// ---------------------------------------------------------------------------
static __device__ inline v16h frag_from(const _Float16* p) {
    h8 lo = *(const h8*)p;
    h8 hi = *(const h8*)(p + 16);
    v16h a;
#pragma unroll
    for (int i = 0; i < 8; ++i) { a[i] = lo[i]; a[i + 8] = hi[i]; }
    return a;
}

// B fragment staged in LDS, lane-major, 16 halfs per lane (32B -> ds_load_b128 x2).
static __device__ inline v16h wfrag(const _Float16* wsm, int frag, int lane) {
    return *(const v16h*)(wsm + frag * 512 + lane * 16);
}

// ---------------------------------------------------------------------------
// Pre-swizzle all 7 weight matrices into WMMA B-fragment order (f16).
// Fragment layout: frag = base + kt*ntiles + nt; element e = lane*16 + i,
// with K = kt*32 + (lane/16)*8 + i (i<8), kt*32 + 16 + (lane/16)*8 + (i-8) (i>=8), N = nt*16 + lane%16. Out-of-range -> 0.
// frag bases: sig0=0(4) sig1=4(8) sig2=12(2) col0=14(4) col1=18(8)
//             col2=26(8) col3=34(2). Total 36 frags * 512 halfs.
// ---------------------------------------------------------------------------
__global__ void prep_weights_kernel(const float* __restrict__ s0, const float* __restrict__ s1,
                                    const float* __restrict__ s2, const float* __restrict__ c0,
                                    const float* __restrict__ c1, const float* __restrict__ c2,
                                    const float* __restrict__ c3, _Float16* __restrict__ wf) {
    int gid = blockIdx.x * 256 + threadIdx.x;
    if (gid >= NFRAGS * 512) return;
    int frag = gid >> 9;
    int e    = gid & 511;
    int lane = e >> 4;
    int i    = e & 15;
    int g    = lane >> 4;
    int nl   = lane & 15;

    const float* W; int kin, nout, ntiles, fb;
    if      (frag <  4) { W = s0; kin = 32; nout = 64; ntiles = 4; fb = 0;  }
    else if (frag < 12) { W = s1; kin = 64; nout = 64; ntiles = 4; fb = 4;  }
    else if (frag < 14) { W = s2; kin = 64; nout = 16; ntiles = 1; fb = 12; }
    else if (frag < 18) { W = c0; kin = 31; nout = 64; ntiles = 4; fb = 14; }
    else if (frag < 26) { W = c1; kin = 64; nout = 64; ntiles = 4; fb = 18; }
    else if (frag < 34) { W = c2; kin = 64; nout = 64; ntiles = 4; fb = 26; }
    else                { W = c3; kin = 64; nout = 3;  ntiles = 1; fb = 34; }

    int rel = frag - fb;
    int kt  = rel / ntiles;
    int nt  = rel % ntiles;
    int K   = kt * 32 + (i < 8 ? g * 8 + i : 16 + g * 8 + (i - 8));
    int N   = nt * 16 + nl;
    float v = (K < kin && N < nout) ? W[K * nout + N] : 0.0f;
    wf[gid] = (_Float16)v;
}

// ---------------------------------------------------------------------------
// Spherical harmonics per ray -> f16 [ray][16]
// ---------------------------------------------------------------------------
__global__ void sh_kernel(const float* __restrict__ rd, _Float16* __restrict__ sh) {
    int ray = blockIdx.x * 256 + threadIdx.x;
    if (ray >= NRAYS) return;
    float x = rd[ray * 3], y = rd[ray * 3 + 1], z = rd[ray * 3 + 2];
    float xx = x * x, yy = y * y, zz = z * z, xy = x * y, yz = y * z, xz = x * z;
    float o[16];
    o[0]  = 0.28209479177387814f;
    o[1]  = -0.48860251190291987f * y;
    o[2]  =  0.48860251190291987f * z;
    o[3]  = -0.48860251190291987f * x;
    o[4]  =  1.0925484305920792f * xy;
    o[5]  = -1.0925484305920792f * yz;
    o[6]  =  0.94617469575756f * zz - 0.31539156525252f;
    o[7]  = -1.0925484305920792f * xz;
    o[8]  =  0.5462742152960396f * (xx - yy);
    o[9]  = -0.5900435899266435f * y * (3.0f * xx - yy);
    o[10] =  2.890611442640554f * xy * z;
    o[11] = -0.4570457994644657f * y * (4.0f * zz - xx - yy);
    o[12] =  0.37317633259011546f * z * (2.0f * zz - 3.0f * xx - 3.0f * yy);
    o[13] = -0.4570457994644657f * x * (4.0f * zz - xx - yy);
    o[14] =  1.445305721320277f * z * (xx - yy);
    o[15] = -0.5900435899266435f * x * (xx - 3.0f * yy);
#pragma unroll
    for (int i = 0; i < 16; ++i) sh[ray * 16 + i] = (_Float16)o[i];
}

// ---------------------------------------------------------------------------
// Pass-1 z values: z = near + 2*bound * s/(NSTEPS-1)
// ---------------------------------------------------------------------------
__global__ void zgen_kernel(const float* __restrict__ ro, float* __restrict__ z) {
    int gid = blockIdx.x * 256 + threadIdx.x;
    if (gid >= NPTS) return;
    int ray = gid >> 6, s = gid & 63;
    float ox = ro[ray * 3], oy = ro[ray * 3 + 1], oz = ro[ray * 3 + 2];
    float near = sqrtf(ox * ox + oy * oy + oz * oz) - BOUND;
    z[gid] = near + 2.0f * BOUND * (float)s / (float)(NSTEPS - 1);
}

// ---------------------------------------------------------------------------
// Fused kernel: stage all 36 pre-swizzled weight fragments (36 KB) into LDS
// once per block, hash-encode 16 points/wave into per-wave LDS tiles, then
// run sigma MLP (32->64->64->16) and color MLP (32->64->64->64->16) via WMMA
// with B operands served from LDS. 256 threads = 8 waves, one tile per wave.
// Static LDS: 36864 (weights) + 18432 (tiles) = 55296 B.
// ---------------------------------------------------------------------------
__launch_bounds__(256)
__global__ void mlp_kernel(const float* __restrict__ ro, const float* __restrict__ rd,
                           const float* __restrict__ table, const _Float16* __restrict__ wf,
                           const _Float16* __restrict__ sh, const float* __restrict__ zb,
                           float* __restrict__ sig_out, float* __restrict__ rgb_out) {
    __shared__ _Float16 wlds[WLDS_HALFS] __attribute__((aligned(16)));
    __shared__ _Float16 smem[8][16][72] __attribute__((aligned(16)));
    const int lane = threadIdx.x & 31;
    const int wave = threadIdx.x >> 5;
    const int tile = blockIdx.x * 8 + wave;
    const int base = tile * 16;
    const int col  = lane & 15;
    const int g    = lane >> 4;
    _Float16 (*act)[72] = smem[wave];

    // ---- cooperative weight staging: 36864 B as 2304 x 16B ----
    {
        const uint4* srcw = (const uint4*)wf;
        uint4* dstw = (uint4*)wlds;
#pragma unroll
        for (int i = 0; i < WLDS_HALFS / 8 / 256; ++i)
            dstw[threadIdx.x + i * 256] = srcw[threadIdx.x + i * 256];
    }

    // ---- hash encode: lane handles point `col`, levels {it*2+g} ----
    {
        const int pt  = base + col;
        const int ray = pt >> 6;
        const float ox = ro[ray * 3], oy = ro[ray * 3 + 1], oz = ro[ray * 3 + 2];
        const float dx = rd[ray * 3], dy = rd[ray * 3 + 1], dz = rd[ray * 3 + 2];
        const float z  = zb[pt];
        float p0 = fminf(fmaxf((ox + dx * z + BOUND) * (0.5f / BOUND), 0.0f), 1.0f);
        float p1 = fminf(fmaxf((oy + dy * z + BOUND) * (0.5f / BOUND), 0.0f), 1.0f);
        float p2 = fminf(fmaxf((oz + dz * z + BOUND) * (0.5f / BOUND), 0.0f), 1.0f);
        for (int it = 0; it < 8; ++it) {
            int lvl = it * 2 + g;
            float res = floorf(exp2f(4.0f + (7.0f / 15.0f) * (float)lvl));
            float xs0 = p0 * res, xs1 = p1 * res, xs2 = p2 * res;
            float fl0 = floorf(xs0), fl1 = floorf(xs1), fl2 = floorf(xs2);
            float f0 = xs0 - fl0, f1 = xs1 - fl1, f2 = xs2 - fl2;
            unsigned c0 = (unsigned)(int)fl0, c1 = (unsigned)(int)fl1, c2 = (unsigned)(int)fl2;
            const float* tb = table + (size_t)lvl * TABLE_SIZE * 2;
            float a0 = 0.0f, a1 = 0.0f;
#pragma unroll
            for (int cn = 0; cn < 8; ++cn) {
                unsigned bi = (cn >> 2) & 1, bj = (cn >> 1) & 1, bk = cn & 1;
                unsigned h = (c0 + bi) * 1u ^ (c1 + bj) * 2654435761u ^ (c2 + bk) * 805459861u;
                unsigned idx = h & (TABLE_SIZE - 1);
                float w = (bi ? f0 : 1.0f - f0) * (bj ? f1 : 1.0f - f1) * (bk ? f2 : 1.0f - f2);
                a0 += w * tb[idx * 2];
                a1 += w * tb[idx * 2 + 1];
            }
            act[col][lvl * 2]     = (_Float16)a0;
            act[col][lvl * 2 + 1] = (_Float16)a1;
        }
    }
    __syncthreads();

    // ---- sigma layer 0: 32 -> 64 ----
    v8f cc[4];
    {
        v16h a0 = frag_from(&act[col][g * 8]);
#pragma unroll
        for (int nt = 0; nt < 4; ++nt) {
            v8f c = {0, 0, 0, 0, 0, 0, 0, 0};
            c = __builtin_amdgcn_wmma_f32_16x16x32_f16(false, a0, false, wfrag(wlds, 0 + nt, lane),
                                                       (short)0, c, false, false);
            cc[nt] = c;
        }
    }
    __syncthreads();
#pragma unroll
    for (int nt = 0; nt < 4; ++nt)
#pragma unroll
        for (int r = 0; r < 8; ++r)
            act[r + 8 * g][nt * 16 + col] = (_Float16)fmaxf(cc[nt][r], 0.0f);
    __syncthreads();

    // ---- sigma layer 1: 64 -> 64 ----
    {
        v16h a0 = frag_from(&act[col][g * 8]);
        v16h a1 = frag_from(&act[col][32 + g * 8]);
#pragma unroll
        for (int nt = 0; nt < 4; ++nt) {
            v8f c = {0, 0, 0, 0, 0, 0, 0, 0};
            c = __builtin_amdgcn_wmma_f32_16x16x32_f16(false, a0, false, wfrag(wlds, 4 + nt, lane),
                                                       (short)0, c, false, false);
            c = __builtin_amdgcn_wmma_f32_16x16x32_f16(false, a1, false, wfrag(wlds, 8 + nt, lane),
                                                       (short)0, c, false, false);
            cc[nt] = c;
        }
    }
    __syncthreads();
#pragma unroll
    for (int nt = 0; nt < 4; ++nt)
#pragma unroll
        for (int r = 0; r < 8; ++r)
            act[r + 8 * g][nt * 16 + col] = (_Float16)fmaxf(cc[nt][r], 0.0f);
    __syncthreads();

    // ---- sigma layer 2: 64 -> 16 (no relu). col 0 = sigma, 1..15 = geo ----
    {
        v16h a0 = frag_from(&act[col][g * 8]);
        v16h a1 = frag_from(&act[col][32 + g * 8]);
        v8f c = {0, 0, 0, 0, 0, 0, 0, 0};
        c = __builtin_amdgcn_wmma_f32_16x16x32_f16(false, a0, false, wfrag(wlds, 12, lane),
                                                   (short)0, c, false, false);
        c = __builtin_amdgcn_wmma_f32_16x16x32_f16(false, a1, false, wfrag(wlds, 13, lane),
                                                   (short)0, c, false, false);
        __syncthreads();
        // build color input rows: [0..15]=SH, [16..30]=geo, [31]=0
        if (col == 0) {
#pragma unroll
            for (int r = 0; r < 8; ++r) {
                sig_out[base + r + 8 * g] = c[r];
                act[r + 8 * g][31] = (_Float16)0.0f;
            }
        } else {
#pragma unroll
            for (int r = 0; r < 8; ++r)
                act[r + 8 * g][15 + col] = (_Float16)c[r];
        }
        {
            int ray = (base + col) >> 6;
            h8 s8 = *(const h8*)(sh + ray * 16 + g * 8);
            *(h8*)&act[col][g * 8] = s8;
        }
    }
    __syncthreads();

    // ---- color layer 0: 32 -> 64 ----
    {
        v16h a0 = frag_from(&act[col][g * 8]);
#pragma unroll
        for (int nt = 0; nt < 4; ++nt) {
            v8f c = {0, 0, 0, 0, 0, 0, 0, 0};
            c = __builtin_amdgcn_wmma_f32_16x16x32_f16(false, a0, false, wfrag(wlds, 14 + nt, lane),
                                                       (short)0, c, false, false);
            cc[nt] = c;
        }
    }
    __syncthreads();
#pragma unroll
    for (int nt = 0; nt < 4; ++nt)
#pragma unroll
        for (int r = 0; r < 8; ++r)
            act[r + 8 * g][nt * 16 + col] = (_Float16)fmaxf(cc[nt][r], 0.0f);
    __syncthreads();

    // ---- color layers 1 and 2: 64 -> 64 ----
#pragma unroll
    for (int layer = 0; layer < 2; ++layer) {
        int fb = 18 + layer * 8;
        v16h a0 = frag_from(&act[col][g * 8]);
        v16h a1 = frag_from(&act[col][32 + g * 8]);
#pragma unroll
        for (int nt = 0; nt < 4; ++nt) {
            v8f c = {0, 0, 0, 0, 0, 0, 0, 0};
            c = __builtin_amdgcn_wmma_f32_16x16x32_f16(false, a0, false, wfrag(wlds, fb + nt, lane),
                                                       (short)0, c, false, false);
            c = __builtin_amdgcn_wmma_f32_16x16x32_f16(false, a1, false, wfrag(wlds, fb + 4 + nt, lane),
                                                       (short)0, c, false, false);
            cc[nt] = c;
        }
        __syncthreads();
#pragma unroll
        for (int nt = 0; nt < 4; ++nt)
#pragma unroll
            for (int r = 0; r < 8; ++r)
                act[r + 8 * g][nt * 16 + col] = (_Float16)fmaxf(cc[nt][r], 0.0f);
        __syncthreads();
    }

    // ---- color layer 3: 64 -> 16 (cols 0..2 valid), sigmoid ----
    {
        v16h a0 = frag_from(&act[col][g * 8]);
        v16h a1 = frag_from(&act[col][32 + g * 8]);
        v8f c = {0, 0, 0, 0, 0, 0, 0, 0};
        c = __builtin_amdgcn_wmma_f32_16x16x32_f16(false, a0, false, wfrag(wlds, 34, lane),
                                                   (short)0, c, false, false);
        c = __builtin_amdgcn_wmma_f32_16x16x32_f16(false, a1, false, wfrag(wlds, 35, lane),
                                                   (short)0, c, false, false);
        if (col < 3) {
#pragma unroll
            for (int r = 0; r < 8; ++r)
                rgb_out[(base + r + 8 * g) * 3 + col] = 1.0f / (1.0f + expf(-c[r]));
        }
    }
}

// ---------------------------------------------------------------------------
// Per-ray PDF importance resampling (sample_pdf). One thread per ray.
// ---------------------------------------------------------------------------
__global__ void pdf_kernel(const float* __restrict__ zb, const float* __restrict__ sg,
                           float* __restrict__ z2) {
    int ray = blockIdx.x * 256 + threadIdx.x;
    if (ray >= NRAYS) return;
    float z[NSTEPS];
    for (int k = 0; k < NSTEPS; ++k) z[k] = zb[ray * NSTEPS + k];

    float w[NSTEPS];
    float T = 1.0f;
    for (int k = 0; k < NSTEPS; ++k) {
        float d = (k < NSTEPS - 1) ? (z[k + 1] - z[k]) : 1e10f;
        float s = fmaxf(sg[ray * NSTEPS + k], 0.0f);
        float a = 1.0f - expf(-d * s);
        w[k] = a * T;
        T *= (1.0f - a + 1e-7f);
    }
    float bins[NSTEPS - 1];
    for (int s = 0; s < NSTEPS - 1; ++s) bins[s] = 0.5f * (z[s] + z[s + 1]);

    float cdf[NSTEPS - 1];
    cdf[0] = 0.0f;
    for (int m = 1; m <= NSTEPS - 2; ++m) cdf[m] = cdf[m - 1] + (w[m] + 1e-5f);
    float inv = 1.0f / cdf[NSTEPS - 2];
    for (int m = 0; m <= NSTEPS - 2; ++m) cdf[m] *= inv;

    for (int s = 0; s < NSTEPS; ++s) {
        float u = ((float)s + 0.5f) / (float)NSTEPS;
        int ind = 0;
        while (ind < NSTEPS - 1 && cdf[ind] <= u) ind++;
        int be = ind - 1; if (be < 0) be = 0;
        int ab = ind;     if (ab > NSTEPS - 2) ab = NSTEPS - 2;
        float cdf0 = cdf[be], cdf1 = cdf[ab];
        float den  = (cdf1 - cdf0 < 1e-5f) ? 1.0f : (cdf1 - cdf0);
        float t    = (u - cdf0) / den;
        z2[ray * NSTEPS + s] = bins[be] + t * (bins[ab] - bins[be]);
    }
}

// ---------------------------------------------------------------------------
// Final composite: streaming two-pointer merge of the two sorted z lists
// (stable: pass-1 wins ties), alpha compositing -> depth + image.
// ---------------------------------------------------------------------------
__global__ void comp_kernel(const float* __restrict__ ro,
                            const float* __restrict__ z1, const float* __restrict__ s1,
                            const float* __restrict__ r1,
                            const float* __restrict__ z2, const float* __restrict__ s2,
                            const float* __restrict__ r2, float* __restrict__ out) {
    int ray = blockIdx.x * 256 + threadIdx.x;
    if (ray >= NRAYS) return;
    float ox = ro[ray * 3], oy = ro[ray * 3 + 1], oz = ro[ray * 3 + 2];
    float near = sqrtf(ox * ox + oy * oy + oz * oz) - BOUND;

    int i = 0, j = 0;
    float T = 1.0f, wsum = 0.0f, dacc = 0.0f, ir = 0.0f, ig = 0.0f, ib = 0.0f;
    for (int k = 0; k < 2 * NSTEPS; ++k) {
        bool t1 = (j >= NSTEPS) || (i < NSTEPS && z1[ray * NSTEPS + i] <= z2[ray * NSTEPS + j]);
        int idx = ray * NSTEPS + (t1 ? i : j);
        float zc = t1 ? z1[idx] : z2[idx];
        float sg = t1 ? s1[idx] : s2[idx];
        const float* rc = t1 ? r1 : r2;
        float cr = rc[idx * 3], cg = rc[idx * 3 + 1], cb = rc[idx * 3 + 2];
        if (t1) i++; else j++;

        float delta;
        if (i >= NSTEPS && j >= NSTEPS) delta = 1e10f;
        else {
            bool n1 = (j >= NSTEPS) || (i < NSTEPS && z1[ray * NSTEPS + i] <= z2[ray * NSTEPS + j]);
            float zn = n1 ? z1[ray * NSTEPS + i] : z2[ray * NSTEPS + j];
            delta = zn - zc;
        }
        float a  = 1.0f - expf(-delta * fmaxf(sg, 0.0f));
        float wg = a * T;
        T *= (1.0f - a + 1e-7f);
        float ori = fminf(fmaxf((zc - near) * 0.5f, 0.0f), 1.0f);
        wsum += wg; dacc += wg * ori;
        ir += wg * cr; ig += wg * cg; ib += wg * cb;
    }
    out[ray] = (dacc + 1e-7f) / (wsum + 1e-7f);
    out[NRAYS + ray * 3 + 0] = ir;
    out[NRAYS + ray * 3 + 1] = ig;
    out[NRAYS + ray * 3 + 2] = ib;
}

// ---------------------------------------------------------------------------
extern "C" void kernel_launch(void* const* d_in, const int* in_sizes, int n_in,
                              void* d_out, int out_size, void* d_ws, size_t ws_size,
                              hipStream_t stream) {
    const float* ro  = (const float*)d_in[0];
    const float* rd  = (const float*)d_in[1];
    const float* tab = (const float*)d_in[2];
    const float* s0  = (const float*)d_in[3];
    const float* s1  = (const float*)d_in[4];
    const float* s2  = (const float*)d_in[5];
    const float* c0  = (const float*)d_in[6];
    const float* c1  = (const float*)d_in[7];
    const float* c2  = (const float*)d_in[8];
    const float* c3  = (const float*)d_in[9];
    (void)in_sizes; (void)n_in; (void)out_size; (void)ws_size;

    char* ws = (char*)d_ws;
    _Float16* wf  = (_Float16*)(ws + 0);         //  36864 B used
    _Float16* sh  = (_Float16*)(ws + 73728);     // 131072 B
    float* z1     = (float*)(ws + 204800);       // 1 MB
    float* sg1    = (float*)(ws + 1253376);      // 1 MB
    float* rgb1   = (float*)(ws + 2301952);      // 3 MB
    float* z2     = (float*)(ws + 5447680);      // 1 MB
    float* sg2    = (float*)(ws + 6496256);      // 1 MB
    float* rgb2   = (float*)(ws + 7544832);      // 3 MB

    prep_weights_kernel<<<72, 256, 0, stream>>>(s0, s1, s2, c0, c1, c2, c3, wf);
    sh_kernel<<<NRAYS / 256, 256, 0, stream>>>(rd, sh);
    zgen_kernel<<<NPTS / 256, 256, 0, stream>>>(ro, z1);
    mlp_kernel<<<NPTS / 16 / 8, 256, 0, stream>>>(ro, rd, tab, wf, sh, z1, sg1, rgb1);
    pdf_kernel<<<NRAYS / 256, 256, 0, stream>>>(z1, sg1, z2);
    mlp_kernel<<<NPTS / 16 / 8, 256, 0, stream>>>(ro, rd, tab, wf, sh, z2, sg2, rgb2);
    comp_kernel<<<NRAYS / 256, 256, 0, stream>>>(ro, z1, sg1, rgb1, z2, sg2, rgb2, (float*)d_out);
}
